// Linear_Int4_16458314678409
// MI455X (gfx1250) — compile-verified, matrix operand order checked by reading, not run on hardware
//
#include <hip/hip_runtime.h>

typedef __attribute__((ext_vector_type(16))) _Float16 v16h;
typedef __attribute__((ext_vector_type(8)))  _Float16 v8h;
typedef __attribute__((ext_vector_type(8)))  float    v8f;

#define IN_F    4096
#define OUT_F   4096
#define GROUPSZ 128
#define BM 128
#define BN 128
#define BK 64
#define PAD 8
#define LDK (BK + PAD)    // 72 halfs/row -> 144 B, 16B aligned
#define EPAD 20           // epilogue scratch row stride (floats): conflict-free b128 reads

// y[m,n] = sum_k x[m,k] * (nib(k,n)*scale[g,n] - zero[g,n]) + bias[n]
__global__ __launch_bounds__(256)
void linear_int4_wmma(const float* __restrict__ x,
                      const int*   __restrict__ qw,      // (IN_F/8, OUT_F), 8 nibbles along K
                      const float* __restrict__ scales,  // (32, OUT_F)
                      const float* __restrict__ zeros,   // (32, OUT_F)
                      const float* __restrict__ bias,    // (OUT_F)
                      float* __restrict__ out,           // (M, OUT_F)
                      int M)
{
    __shared__ _Float16 xs[BM][LDK];      // A tile, m-major f16
    __shared__ _Float16 ws[BN][LDK];      // B tile, n-major dequantized f16
    __shared__ float    esc[8][16*EPAD];  // per-wave epilogue transpose scratch

    const int t    = threadIdx.x;
    const int lane = t & 31;
    const int wave = t >> 5;
    const int wm   = wave >> 1;        // 0..3 : 32-row M strip
    const int wn   = wave & 1;         // 0..1 : 64-col N strip
    const int lr   = lane & 15;        // row/col within a 16x16 fragment
    const int kh   = lane >> 4;        // K-half select (WMMA 16-bit layout)

    const int n0 = blockIdx.x * BN;
    const int m0 = blockIdx.y * BM;

    // staging work split
    const int xr   = t >> 1;           // x row 0..127
    const int xh   = t & 1;            // which 32-float half of the row
    const int wcol = t & 127;          // n within tile for dequant
    const int wks  = t >> 7;           // 0/1 : which 4 int32 rows

    v8f acc[2][4];
    #pragma unroll
    for (int i = 0; i < 2; ++i)
        #pragma unroll
        for (int j = 0; j < 4; ++j)
            acc[i][j] = (v8f){0.f,0.f,0.f,0.f,0.f,0.f,0.f,0.f};

    // ---- register staging buffers (pipeline depth 1) ----
    float4 xraw[8];
    int    wraw[4];
    float  sreg, zreg;

    auto loadTile = [&](int k0) {
        const float4* xp =
            (const float4*)(x + (size_t)(m0 + xr) * IN_F + k0 + xh * 32);
        #pragma unroll
        for (int i = 0; i < 8; ++i) xraw[i] = xp[i];
        const int g = k0 >> 7;                           // GROUP = 128
        sreg = scales[(size_t)g * OUT_F + n0 + wcol];
        zreg = zeros [(size_t)g * OUT_F + n0 + wcol];
        #pragma unroll
        for (int q = 0; q < 4; ++q)
            wraw[q] = qw[(size_t)(k0/8 + wks*4 + q) * OUT_F + n0 + wcol];
    };

    auto storeTile = [&]() {
        // X: fp32 -> f16 into LDS (m-major)
        #pragma unroll
        for (int i = 0; i < 4; ++i) {
            float4 f0 = xraw[2*i + 0];
            float4 f1 = xraw[2*i + 1];
            v8h h;
            h[0]=(_Float16)f0.x; h[1]=(_Float16)f0.y;
            h[2]=(_Float16)f0.z; h[3]=(_Float16)f0.w;
            h[4]=(_Float16)f1.x; h[5]=(_Float16)f1.y;
            h[6]=(_Float16)f1.z; h[7]=(_Float16)f1.w;
            *(v8h*)&xs[xr][xh*32 + i*8] = h;
        }
        // W: unpack int4 nibbles, dequant, store n-major
        #pragma unroll
        for (int q = 0; q < 4; ++q) {
            const int kw   = wks * 4 + q;
            const int word = wraw[q];
            v8h h;
            #pragma unroll
            for (int j = 0; j < 8; ++j) {
                const int nib = (word >> (4*j)) & 0xF;
                h[j] = (_Float16)((float)nib * sreg - zreg);
            }
            *(v8h*)&ws[wcol][kw*8] = h;
        }
    };

    loadTile(0);

    for (int k0 = 0; k0 < IN_F; k0 += BK) {
        storeTile();
        __syncthreads();

        // prefetch next tile's globals; they fly during the WMMAs below
        if (k0 + BK < IN_F) loadTile(k0 + BK);

        #pragma unroll
        for (int kt = 0; kt < BK; kt += 32) {
            v16h a[2], b[4];
            #pragma unroll
            for (int i = 0; i < 2; ++i) {
                const int row = wm*32 + i*16 + lr;
                *((v8h*)&a[i] + 0) = *(const v8h*)&xs[row][kt +      kh*8];
                *((v8h*)&a[i] + 1) = *(const v8h*)&xs[row][kt + 16 + kh*8];
            }
            #pragma unroll
            for (int j = 0; j < 4; ++j) {
                const int col = wn*64 + j*16 + lr;
                *((v8h*)&b[j] + 0) = *(const v8h*)&ws[col][kt +      kh*8];
                *((v8h*)&b[j] + 1) = *(const v8h*)&ws[col][kt + 16 + kh*8];
            }
            #pragma unroll
            for (int i = 0; i < 2; ++i)
                #pragma unroll
                for (int j = 0; j < 4; ++j)
                    acc[i][j] = __builtin_amdgcn_wmma_f32_16x16x32_f16(
                        false, a[i], false, b[j],
                        (short)0, acc[i][j], false, false);
        }
        __syncthreads();
    }

    // ---- epilogue: transpose each 16x16 C fragment via LDS, add bias,
    //      store 32B/lane contiguous (global_store_b128) ----
    float* sc = esc[wave];
    #pragma unroll
    for (int i = 0; i < 2; ++i) {
        #pragma unroll
        for (int j = 0; j < 4; ++j) {
            __syncthreads();
            // acc VGPR r holds element (mm = kh*8 + r, nn = lr)
            #pragma unroll
            for (int r = 0; r < 8; ++r)
                sc[(kh*8 + r)*EPAD + lr] = acc[i][j][r];
            __syncthreads();
            // lane reads row mm = lr, column half kh -> 8 contiguous floats
            const float* srow = &sc[lr*EPAD + kh*8];
            float4 c0 = *(const float4*)(srow + 0);
            float4 c1 = *(const float4*)(srow + 4);
            const int nb = n0 + wn*64 + j*16 + kh*8;
            float4 b0 = *(const float4*)&bias[nb];
            float4 b1 = *(const float4*)&bias[nb + 4];
            c0.x += b0.x; c0.y += b0.y; c0.z += b0.z; c0.w += b0.w;
            c1.x += b1.x; c1.y += b1.y; c1.z += b1.z; c1.w += b1.w;
            const int m = m0 + wm*32 + i*16 + lr;
            float* op = out + (size_t)m * OUT_F + nb;
            *(float4*)(op + 0) = c0;
            *(float4*)(op + 4) = c1;
        }
    }
}

extern "C" void kernel_launch(void* const* d_in, const int* in_sizes, int n_in,
                              void* d_out, int out_size, void* d_ws, size_t ws_size,
                              hipStream_t stream) {
    const float* x      = (const float*)d_in[0];
    const int*   qw     = (const int*)  d_in[1];
    const float* scales = (const float*)d_in[2];
    const float* zeros  = (const float*)d_in[3];
    const float* bias   = (const float*)d_in[4];
    float* out = (float*)d_out;

    const int M = in_sizes[0] / IN_F;       // 4*2048 = 8192
    dim3 grid(OUT_F / BN, M / BM);          // (32, 64)
    linear_int4_wmma<<<grid, 256, 0, stream>>>(x, qw, scales, zeros, bias, out, M);
}
